// GenReV3_35897336660033
// MI455X (gfx1250) — matrix operand order repaired and checked by reading, not compiled, not run on hardware
//
#include <hip/hip_runtime.h>
#include <hip/hip_bf16.h>
#include <math.h>

// ---------------- model constants ----------------
#define BSZ   64
#define DM    256
#define FFD   1024
#define NLAY  6
#define SFEAT 48      // S
#define VOC   1024
#define NH    8
#define DH    32
#define NCONT 32
#define NCAT  16
// output layout: s_cont [64,32] | s_cat [64,16] | logits [48,64,1024]
#define OUT_LOGITS_BASE (BSZ*NCONT + BSZ*NCAT)   // 3072

// f16 weight arena (per-layer offsets, in halfs)
#define HL_SW     0         // dec_swqkv   [768,256]
#define HL_SWO    196608    // dec_swo     [256,256]
#define HL_CQ     262144    // dec_cwqkv rows 0..255 (q part)
#define HL_CWO    327680    // dec_cwo     [256,256]
#define HL_W1     393216    // dec_w1      [1024,256]
#define HL_W2     655360    // dec_w2      [256,1024]
#define HL_STRIDE 917504

typedef __attribute__((ext_vector_type(16))) _Float16 v16h;
typedef __attribute__((ext_vector_type(8)))  float    v8f;

// ---------------- helpers ----------------
__device__ __forceinline__ float pos_enc(int p, int d) {
  int i2 = d & ~1;
  float div = __expf((float)i2 * (-9.2103403719761836f / 256.0f));
  float ang = (float)p * div;
  return (d & 1) ? __cosf(ang) : __sinf(ang);
}

__device__ __forceinline__ float hash_u01(unsigned x) {
  x ^= x >> 17; x *= 0xed5ad4bbu;
  x ^= x >> 11; x *= 0xac4c1b51u;
  x ^= x >> 15; x *= 0x31848babu;
  x ^= x >> 14;
  return ((float)(x >> 8) + 0.5f) * (1.0f / 16777216.0f);
}

// B fragment: 16 contiguous K values for one N column
__device__ __forceinline__ v16h load_bfrag(const _Float16* __restrict__ p) {
  return *(const v16h*)p;                       // 32B direct f16 load, no converts
}
__device__ __forceinline__ v16h load_bfrag(const float* __restrict__ p) {
  v8f lo = *(const v8f*)p;
  v8f hi = *(const v8f*)(p + 8);
  v16h b;
#pragma unroll
  for (int h = 0; h < 8; ++h) { b[h] = (_Float16)lo[h]; b[h + 8] = (_Float16)hi[h]; }
  return b;
}
// A fragment: halves 0..15 <- p[0..15] (lane-dependent base)
__device__ __forceinline__ v16h load_afrag(const float* __restrict__ p) {
  v8f lo = *(const v8f*)p;
  v8f hi = *(const v8f*)(p + 8);
  v16h a;
#pragma unroll
  for (int h = 0; h < 8; ++h) { a[h] = (_Float16)lo[h]; a[h + 8] = (_Float16)hi[h]; }
  return a;
}

// ---------------- WMMA GEMM: Y[M,N] = act(X[M,K] @ W[N,K]^T + bias) ----------------
// One wave computes a 32(M)x64(N) tile: 2 A frags x 4 B frags -> 8 v_wmma per
// K-step; all fragment loads grouped before the WMMA cluster. 8 waves/block
// arranged 2(M)x4(N) cover a 64x256 macro-tile. All M here are multiples of 64,
// all N of 256. W is templated: f16 (pre-converted, hot decode path) or f32.
// A (16x32 f16) lanes 0-15: K=0..15; lanes 16-31: K=16..31
// B (32x16 f16) lanes 0-15: K=0..15; lanes 16-31: K=16..31 (col n = lane%16)
// C/D (16x16 f32): vgpr r -> row r (lanes 0-15) / r+8 (lanes 16-31), col = lane%16
template <typename WT>
__global__ __launch_bounds__(256) void gemm_wmma(
    const float* __restrict__ X, const WT* __restrict__ W,
    const float* __restrict__ bias, float* __restrict__ Y,
    int M, int N, int K, int relu)
{
  const int wave   = threadIdx.x >> 5;
  const int lane   = threadIdx.x & 31;
  const int tileM  = blockIdx.y * 64 + (wave >> 2) * 32;
  const int tileN0 = blockIdx.x * 256 + (wave & 3) * 64;
  if (tileM >= M || tileN0 >= N) return;   // wave-uniform (grids exact anyway)

  const int nbase  = tileN0 + (lane & 15);
  const int kaBase = (lane < 16) ? 0 : 16;
  const int kbBase = (lane < 16) ? 0 : 16;
  const float* __restrict__ xrow0 = X + (size_t)(tileM + (lane & 15)) * K;
  const float* __restrict__ xrow1 = xrow0 + (size_t)16 * K;
  const WT*    __restrict__ wrow  = W + (size_t)nbase * K;

  v8f acc[2][4];
#pragma unroll
  for (int g = 0; g < 2; ++g)
#pragma unroll
    for (int s = 0; s < 4; ++s)
#pragma unroll
      for (int r = 0; r < 8; ++r) acc[g][s][r] = 0.0f;

  for (int kk = 0; kk < K; kk += 32) {
    if (kk + 64 < K) {
      __builtin_prefetch(xrow0 + kk + 64, 0, 1);              // global_prefetch_b8
      __builtin_prefetch(wrow + kk + 64, 0, 1);
    }
    // gather all fragments first so the loads can overlap under one wait
    v16h a0 = load_afrag(xrow0 + kk + kaBase);
    v16h a1 = load_afrag(xrow1 + kk + kaBase);
    v16h b[4];
#pragma unroll
    for (int s = 0; s < 4; ++s)
      b[s] = load_bfrag(wrow + (size_t)(s * 16) * K + kk + kbBase);
#pragma unroll
    for (int s = 0; s < 4; ++s)
      acc[0][s] = __builtin_amdgcn_wmma_f32_16x16x32_f16(
          false, a0, false, b[s], (short)0, acc[0][s], false, false);
#pragma unroll
    for (int s = 0; s < 4; ++s)
      acc[1][s] = __builtin_amdgcn_wmma_f32_16x16x32_f16(
          false, a1, false, b[s], (short)0, acc[1][s], false, false);
  }

#pragma unroll
  for (int g = 0; g < 2; ++g) {
    const int mrow = tileM + g * 16 + ((lane < 16) ? 0 : 8);
#pragma unroll
    for (int s = 0; s < 4; ++s) {
      const int   n  = nbase + s * 16;
      const float bv = bias ? bias[n] : 0.0f;
#pragma unroll
      for (int r = 0; r < 8; ++r) {
        float v = acc[g][s][r] + bv;
        if (relu) v = fmaxf(v, 0.0f);
        Y[(size_t)(mrow + r) * N + n] = v;
      }
    }
  }
}

// ---------------- fp32 -> f16 weight conversion (once per launch) ----------------
__global__ __launch_bounds__(256) void cvt_f16_kernel(
    const float* __restrict__ src, _Float16* __restrict__ dst, int n)
{
  int i = blockIdx.x * 256 + threadIdx.x;
  if (i < n) dst[i] = (_Float16)src[i];
}

// ---------------- LayerNorm (optionally fused residual add) ----------------
__global__ __launch_bounds__(256) void ln_kernel(
    const float* __restrict__ x, const float* __restrict__ add,
    const float* __restrict__ w, const float* __restrict__ b,
    float* __restrict__ y, int rowoff, int nrows)
{
  const int lane = threadIdx.x & 31;
  const int wid  = threadIdx.x >> 5;
  const int row  = blockIdx.x * 8 + wid;
  if (row >= nrows) return;
  const size_t rbase = (size_t)(rowoff + row) * DM;
  float v[8];
  float s = 0.0f;
#pragma unroll
  for (int i = 0; i < 8; ++i) {
    float t = x[rbase + lane + i * 32];
    if (add) t += add[(size_t)row * DM + lane + i * 32];
    v[i] = t; s += t;
  }
#pragma unroll
  for (int off = 16; off > 0; off >>= 1) s += __shfl_xor(s, off, 32);
  const float mean = s * (1.0f / 256.0f);
  float var = 0.0f;
#pragma unroll
  for (int i = 0; i < 8; ++i) { float d = v[i] - mean; var += d * d; }
#pragma unroll
  for (int off = 16; off > 0; off >>= 1) var += __shfl_xor(var, off, 32);
  const float rs = rsqrtf(var * (1.0f / 256.0f) + 1e-5f);
#pragma unroll
  for (int i = 0; i < 8; ++i) {
    int d = lane + i * 32;
    y[(size_t)row * DM + d] = (v[i] - mean) * rs * w[d] + b[d];
  }
}

// ---------------- attention (flash-style online softmax, one block per (b,h)) ----------------
__global__ __launch_bounds__(64) void attn_kernel(
    const float* __restrict__ q, int ldq, int qoff,
    const float* __restrict__ kv, int ldkv, int koff, int voff,
    float* __restrict__ o, int Lq, int Lkv, int causal, float scale)
{
  __shared__ float sk[(SFEAT + 1) * DH];
  __shared__ float sv[(SFEAT + 1) * DH];
  const int b = blockIdx.x, h = blockIdx.y, tid = threadIdx.x;
  for (int idx = tid; idx < Lkv * DH; idx += 64) {
    int j = idx >> 5, d = idx & 31;
    size_t base = (size_t)(j * BSZ + b) * ldkv + h * DH + d;
    sk[idx] = kv[base + koff];
    sv[idx] = kv[base + voff];
  }
  __syncthreads();
  const int p = tid;
  if (p >= Lq) return;

  float qv[DH];
  const float* qr = q + (size_t)(p * BSZ + b) * ldq + qoff + h * DH;
#pragma unroll
  for (int d = 0; d < DH; ++d) qv[d] = qr[d];

  float mm = -1e30f, l = 0.0f, acc[DH];
#pragma unroll
  for (int d = 0; d < DH; ++d) acc[d] = 0.0f;

  const int jend = causal ? (p + 1) : Lkv;
  for (int j = 0; j < jend; ++j) {
    float s = 0.0f;
#pragma unroll
    for (int d = 0; d < DH; ++d) s += qv[d] * sk[j * DH + d];
    s *= scale;
    float nm = fmaxf(mm, s);
    float corr = __expf(mm - nm);
    float wgt  = __expf(s - nm);
    l = l * corr + wgt;
#pragma unroll
    for (int d = 0; d < DH; ++d) acc[d] = acc[d] * corr + wgt * sv[j * DH + d];
    mm = nm;
  }
  const float inv = 1.0f / l;
  float* orow = o + (size_t)(p * BSZ + b) * DM + h * DH;
#pragma unroll
  for (int d = 0; d < DH; ++d) orow[d] = acc[d] * inv;
}

// ---------------- encoder embedding: x = emb_tab[s, idx[b,s]] + pe[s] ----------------
__global__ __launch_bounds__(256) void embed_enc_kernel(
    const int* __restrict__ xc, const int* __restrict__ xk,
    const float* __restrict__ emb, float* __restrict__ X0)
{
  int idx = blockIdx.x * 256 + threadIdx.x;
  int d = idx & 255, t = idx >> 8;
  int s = t >> 6, b = t & 63;
  int tok = (s < NCONT) ? xc[b * NCONT + s] : xk[b * NCAT + (s - NCONT)];
  X0[(size_t)t * DM + d] = emb[((size_t)s * VOC + tok) * DM + d] + pos_enc(s, d);
}

// ---------------- decoder input: X0 = buf + pe ----------------
__global__ __launch_bounds__(256) void dec_embed_kernel(
    const float* __restrict__ buf, float* __restrict__ X0)
{
  int idx = blockIdx.x * 256 + threadIdx.x;
  int d = idx & 255, t = idx >> 8, p = t >> 6;
  X0[idx] = buf[idx] + pos_enc(p, d);
}

// ---------------- decoder buffer init: buf[p=0][b] = sos ----------------
__global__ __launch_bounds__(256) void initbuf_kernel(
    const float* __restrict__ sos, float* __restrict__ buf)
{
  buf[(size_t)blockIdx.x * DM + threadIdx.x] = sos[threadIdx.x];
}

// ---------------- gumbel argmax sample + write choice + append embedding ----------------
__global__ __launch_bounds__(256) void sample_kernel(
    const float* __restrict__ logits, const float* __restrict__ emb,
    float* __restrict__ buf, float* __restrict__ out, int step)
{
  __shared__ float sval[256];
  __shared__ int   sidx[256];
  __shared__ int   schoice;
  const int b = blockIdx.x, tid = threadIdx.x;
  float best = -1e30f; int bi = 0;
  for (int v = tid; v < VOC; v += 256) {
    unsigned key = (unsigned)(((step * BSZ + b) << 10) | v) * 2654435761u + 0x9E3779B9u;
    float u = hash_u01(key);
    float g = -__logf(-__logf(u));
    float val = logits[b * VOC + v] * 10.0f + g;   // /TEMP, TEMP=0.1
    if (val > best) { best = val; bi = v; }
  }
  sval[tid] = best; sidx[tid] = bi;
  __syncthreads();
  for (int s = 128; s > 0; s >>= 1) {
    if (tid < s && sval[tid + s] > sval[tid]) { sval[tid] = sval[tid + s]; sidx[tid] = sidx[tid + s]; }
    __syncthreads();
  }
  if (tid == 0) {
    schoice = sidx[0];
    float c = (float)sidx[0];
    if (step < NCONT) out[b * NCONT + step] = c;
    else              out[BSZ * NCONT + b * NCAT + (step - NCONT)] = c;
  }
  __syncthreads();
  int choice = schoice;
  buf[((size_t)(step + 1) * BSZ + b) * DM + tid] =
      emb[((size_t)step * VOC + choice) * DM + tid];
}

// ---------------- host orchestration ----------------
extern "C" void kernel_launch(void* const* d_in, const int* in_sizes, int n_in,
                              void* d_out, int out_size, void* d_ws, size_t ws_size,
                              hipStream_t stream) {
  (void)in_sizes; (void)n_in; (void)out_size; (void)ws_size;
  const int*   x_cont   = (const int*)  d_in[0];
  const int*   x_cat    = (const int*)  d_in[1];
  const float* sos      = (const float*)d_in[2];
  const float* emb_tab  = (const float*)d_in[3];
  const float* head_w   = (const float*)d_in[4];
  const float* head_b   = (const float*)d_in[5];
  const float* enc_wqkv = (const float*)d_in[6];
  const float* enc_bqkv = (const float*)d_in[7];
  const float* enc_wo   = (const float*)d_in[8];
  const float* enc_bo   = (const float*)d_in[9];
  const float* enc_w1   = (const float*)d_in[10];
  const float* enc_b1   = (const float*)d_in[11];
  const float* enc_w2   = (const float*)d_in[12];
  const float* enc_b2   = (const float*)d_in[13];
  const float* enc_lnw  = (const float*)d_in[14];
  const float* enc_lnb  = (const float*)d_in[15];
  const float* dec_swqkv= (const float*)d_in[16];
  const float* dec_sbqkv= (const float*)d_in[17];
  const float* dec_swo  = (const float*)d_in[18];
  const float* dec_sbo  = (const float*)d_in[19];
  const float* dec_cwqkv= (const float*)d_in[20];
  const float* dec_cbqkv= (const float*)d_in[21];
  const float* dec_cwo  = (const float*)d_in[22];
  const float* dec_cbo  = (const float*)d_in[23];
  const float* dec_w1   = (const float*)d_in[24];
  const float* dec_b1   = (const float*)d_in[25];
  const float* dec_w2   = (const float*)d_in[26];
  const float* dec_b2   = (const float*)d_in[27];
  const float* dec_lnw  = (const float*)d_in[28];
  const float* dec_lnb  = (const float*)d_in[29];
  const float* lnf_w    = (const float*)d_in[30];
  const float* lnf_b    = (const float*)d_in[31];
  float* out = (float*)d_out;

  // workspace layout (floats); max token rows = 48*64 = 3072
  float* ws  = (float*)d_ws;
  float* X0  = ws;                       // 3072*256
  float* X1  = X0  + 3072 * 256;
  float* X2  = X1  + 3072 * 256;
  float* QKV = X2  + 3072 * 256;         // 3072*768
  float* FFH = QKV + 3072 * 768;         // 3072*1024
  float* MEM = FFH + 3072 * 1024;        // 3072*256
  float* BUF = MEM + 3072 * 256;         // 49*64*256
  float* CKV = BUF + 3136 * 256;         // 6 * 3072 * 512
  _Float16* WH = (_Float16*)(CKV + (size_t)6 * 3072 * 512);   // f16 weight arena

  const float scale = 0.17677669529663687f;   // 1/sqrt(32)

  // ---------------- pre-convert decode-loop weights to f16 (once per launch) ----------------
  for (int l = 0; l < NLAY; ++l) {
    _Float16* base = WH + (size_t)l * HL_STRIDE;
    cvt_f16_kernel<<<196608 / 256, 256, 0, stream>>>(dec_swqkv + (size_t)l * 196608, base + HL_SW,  196608);
    cvt_f16_kernel<<< 65536 / 256, 256, 0, stream>>>(dec_swo   + (size_t)l *  65536, base + HL_SWO,  65536);
    cvt_f16_kernel<<< 65536 / 256, 256, 0, stream>>>(dec_cwqkv + (size_t)l * 196608, base + HL_CQ,   65536);
    cvt_f16_kernel<<< 65536 / 256, 256, 0, stream>>>(dec_cwo   + (size_t)l *  65536, base + HL_CWO,  65536);
    cvt_f16_kernel<<<262144 / 256, 256, 0, stream>>>(dec_w1    + (size_t)l * 262144, base + HL_W1,  262144);
    cvt_f16_kernel<<<262144 / 256, 256, 0, stream>>>(dec_w2    + (size_t)l * 262144, base + HL_W2,  262144);
  }

  // ---------------- encoder (M = 3072 tokens, p-major t = s*64+b), fp32 weights ----------------
  const int ME = SFEAT * BSZ;
  embed_enc_kernel<<<ME, 256, 0, stream>>>(x_cont, x_cat, emb_tab, X0);
  for (int l = 0; l < NLAY; ++l) {
    gemm_wmma<<<dim3(3, ME / 64), 256, 0, stream>>>(
        X0, enc_wqkv + (size_t)l * 768 * 256, enc_bqkv + l * 768, QKV, ME, 768, 256, 0);
    attn_kernel<<<dim3(BSZ, NH), 64, 0, stream>>>(
        QKV, 768, 0, QKV, 768, 256, 512, X1, SFEAT, SFEAT, 0, scale);
    gemm_wmma<<<dim3(1, ME / 64), 256, 0, stream>>>(
        X1, enc_wo + (size_t)l * 256 * 256, enc_bo + l * 256, X2, ME, 256, 256, 0);
    ln_kernel<<<ME / 8, 256, 0, stream>>>(
        X0, X2, enc_lnw + (l * 2 + 0) * 256, enc_lnb + (l * 2 + 0) * 256, X0, 0, ME);
    gemm_wmma<<<dim3(4, ME / 64), 256, 0, stream>>>(
        X0, enc_w1 + (size_t)l * 1024 * 256, enc_b1 + l * 1024, FFH, ME, 1024, 256, 1);
    gemm_wmma<<<dim3(1, ME / 64), 256, 0, stream>>>(
        FFH, enc_w2 + (size_t)l * 256 * 1024, enc_b2 + l * 256, X2, ME, 256, 1024, 0);
    ln_kernel<<<ME / 8, 256, 0, stream>>>(
        X0, X2, enc_lnw + (l * 2 + 1) * 256, enc_lnb + (l * 2 + 1) * 256, X0, 0, ME);
  }
  ln_kernel<<<ME / 8, 256, 0, stream>>>(X0, nullptr, lnf_w, lnf_b, MEM, 0, ME);

  // ---------------- cross-attn K/V: fixed per layer, precompute once (fp32 weights) ----------------
  for (int l = 0; l < NLAY; ++l) {
    gemm_wmma<<<dim3(2, ME / 64), 256, 0, stream>>>(
        MEM, dec_cwqkv + (size_t)l * 768 * 256 + 256 * 256, dec_cbqkv + l * 768 + 256,
        CKV + (size_t)l * 3072 * 512, ME, 512, 256, 0);
  }

  // ---------------- autoregressive decode: 48 steps, causal prefix only ----------------
  initbuf_kernel<<<BSZ, 256, 0, stream>>>(sos, BUF);
  for (int i = 0; i < SFEAT; ++i) {
    const int M  = (i + 1) * BSZ;                // only positions p <= i matter
    const int GY = M / 64;
    const _Float16* WB = WH;                     // f16 arena
    dec_embed_kernel<<<M, 256, 0, stream>>>(BUF, X0);
    for (int l = 0; l < NLAY; ++l) {
      const _Float16* LB = WB + (size_t)l * HL_STRIDE;
      // self-attention (causal)
      gemm_wmma<<<dim3(3, GY), 256, 0, stream>>>(
          X0, LB + HL_SW, dec_sbqkv + l * 768, QKV, M, 768, 256, 0);
      attn_kernel<<<dim3(BSZ, NH), 64, 0, stream>>>(
          QKV, 768, 0, QKV, 768, 256, 512, X1, i + 1, i + 1, 1, scale);
      gemm_wmma<<<dim3(1, GY), 256, 0, stream>>>(
          X1, LB + HL_SWO, dec_sbo + l * 256, X2, M, 256, 256, 0);
      ln_kernel<<<M / 8, 256, 0, stream>>>(
          X0, X2, dec_lnw + (l * 3 + 0) * 256, dec_lnb + (l * 3 + 0) * 256, X0, 0, M);
      // cross-attention (q only; K/V precomputed)
      gemm_wmma<<<dim3(1, GY), 256, 0, stream>>>(
          X0, LB + HL_CQ, dec_cbqkv + l * 768, QKV, M, 256, 256, 0);
      attn_kernel<<<dim3(BSZ, NH), 64, 0, stream>>>(
          QKV, 256, 0, CKV + (size_t)l * 3072 * 512, 512, 0, 256, X1, i + 1, SFEAT, 0, scale);
      gemm_wmma<<<dim3(1, GY), 256, 0, stream>>>(
          X1, LB + HL_CWO, dec_cbo + l * 256, X2, M, 256, 256, 0);
      ln_kernel<<<M / 8, 256, 0, stream>>>(
          X0, X2, dec_lnw + (l * 3 + 1) * 256, dec_lnb + (l * 3 + 1) * 256, X0, 0, M);
      // FFN
      gemm_wmma<<<dim3(4, GY), 256, 0, stream>>>(
          X0, LB + HL_W1, dec_b1 + l * 1024, FFH, M, 1024, 256, 1);
      gemm_wmma<<<dim3(1, GY), 256, 0, stream>>>(
          FFH, LB + HL_W2, dec_b2 + l * 256, X2, M, 256, 1024, 0);
      ln_kernel<<<M / 8, 256, 0, stream>>>(
          X0, X2, dec_lnw + (l * 3 + 2) * 256, dec_lnb + (l * 3 + 2) * 256, X0, 0, M);
    }
    // final LN only on position p == i (rows i*64 .. i*64+63) -> X1[0..63]
    ln_kernel<<<8, 256, 0, stream>>>(X0, nullptr, lnf_w + 256, lnf_b + 256, X1, i * BSZ, BSZ);
    // logits for this step, written straight into d_out's logits_all[i] (fp32 head weights)
    float* logit_i = out + OUT_LOGITS_BASE + (size_t)i * BSZ * VOC;
    gemm_wmma<<<dim3(4, 1), 256, 0, stream>>>(
        X1, head_w + (size_t)i * VOC * 256, head_b + (size_t)i * VOC, logit_i, BSZ, VOC, 256, 0);
    // sample + write choice + append next-token embedding to BUF
    sample_kernel<<<BSZ, 256, 0, stream>>>(logit_i, emb_tab, BUF, out, i);
  }
}
